// TernaryMVAdapter_42064909697519
// MI455X (gfx1250) — 8-bit matrix operand order repaired and checked by reading, compiled, not run on hardware
//
#include <hip/hip_runtime.h>
#include <hip/hip_bf16.h>
#include <math.h>

#define HDIM   1152
#define LTOK   256
#define BNIMG  16
#define NVIEW  4
#define NTOK   4096
#define INTERD 4608
#define HEADD  72
#define NHEAD  16

typedef int v8i __attribute__((ext_vector_type(8)));

#if defined(__HIP_DEVICE_COMPILE__) && __has_builtin(__builtin_amdgcn_global_load_async_to_lds_b64)
#define TMVA_ASYNC 1
#else
#define TMVA_ASYNC 0
#endif

typedef int tmva_v2i __attribute__((vector_size(2 * sizeof(int))));

__device__ __forceinline__ void tmva_cp8(const signed char* g, signed char* l) {
#if TMVA_ASYNC
  typedef __attribute__((address_space(1))) tmva_v2i gvec_t;
  typedef __attribute__((address_space(3))) tmva_v2i lvec_t;
  __builtin_amdgcn_global_load_async_to_lds_b64((gvec_t*)g, (lvec_t*)l, 0, 0);
#else
  *(long long*)l = *(const long long*)g;
#endif
}

__device__ __forceinline__ void tmva_cpwait() {
#if TMVA_ASYNC
#if __has_builtin(__builtin_amdgcn_s_wait_asynccnt)
  __builtin_amdgcn_s_wait_asynccnt(0);
#else
  asm volatile("s_wait_asynccnt 0x0" ::: "memory");
#endif
#endif
}

__global__ void tmva_wred1(const float* __restrict__ w, long n, float* __restrict__ part) {
  __shared__ float sm[256];
  long i = (long)blockIdx.x * 256 + threadIdx.x;
  long stride = (long)gridDim.x * 256;
  float s = 0.f;
  for (; i < n; i += stride) s += fabsf(w[i]);
  sm[threadIdx.x] = s; __syncthreads();
  for (int o = 128; o > 0; o >>= 1) {
    if ((int)threadIdx.x < o) sm[threadIdx.x] += sm[threadIdx.x + o];
    __syncthreads();
  }
  if (threadIdx.x == 0) part[blockIdx.x] = sm[0];
}

__global__ void tmva_wred2(const float* __restrict__ part, int nb, long n, float* __restrict__ wslot) {
  __shared__ float sm[256];
  float s = 0.f;
  for (int i = threadIdx.x; i < nb; i += 256) s += part[i];
  sm[threadIdx.x] = s; __syncthreads();
  for (int o = 128; o > 0; o >>= 1) {
    if ((int)threadIdx.x < o) sm[threadIdx.x] += sm[threadIdx.x + o];
    __syncthreads();
  }
  if (threadIdx.x == 0) {
    float mean = sm[0] / (float)n;
    float wd = fmaxf(mean, 1e-5f);
    wslot[0] = wd;
    wslot[1] = 1.0f / wd;
  }
}

__global__ void tmva_wquant(const float* __restrict__ w, signed char* __restrict__ q,
                            long n, const float* __restrict__ wslot) {
  float s = wslot[1];
  long i = (long)blockIdx.x * 256 + threadIdx.x;
  long stride = (long)gridDim.x * 256;
  for (; i < n; i += stride) {
    float v = rintf(w[i] * s);
    v = fminf(1.f, fmaxf(-1.f, v));
    q[i] = (signed char)(int)v;
  }
}

__global__ void tmva_aquant(const float* __restrict__ x, signed char* __restrict__ q,
                            float* __restrict__ ascale, int K, int perm) {
  int r = blockIdx.x;
  long src = r;
  if (perm) {
    int bl = r >> 2, v = r & 3;
    int b = bl >> 8, l = bl & 255;
    src = (long)((b * NVIEW + v) * LTOK + l);
  }
  const float* xr = x + src * (long)K;
  __shared__ float ssum[256];
  __shared__ float smax[256];
  float s = 0.f, mx = 0.f;
  for (int k = threadIdx.x; k < K; k += 256) {
    float v = xr[k]; s += v * v; mx = fmaxf(mx, fabsf(v));
  }
  ssum[threadIdx.x] = s; smax[threadIdx.x] = mx; __syncthreads();
  for (int o = 128; o > 0; o >>= 1) {
    if ((int)threadIdx.x < o) {
      ssum[threadIdx.x] += ssum[threadIdx.x + o];
      smax[threadIdx.x] = fmaxf(smax[threadIdx.x], smax[threadIdx.x + o]);
    }
    __syncthreads();
  }
  __shared__ float sfac;
  if (threadIdx.x == 0) {
    float rms = rsqrtf(ssum[0] / (float)K + 1e-8f);
    float mxn = fmaxf(smax[0] * rms, 1e-5f);
    sfac = rms * (127.f / mxn);
    ascale[r] = mxn / 127.f;
  }
  __syncthreads();
  float f = sfac;
  signed char* qr = q + (long)r * K;
  for (int k = threadIdx.x; k < K; k += 256) {
    float v = rintf(xr[k] * f);
    v = fminf(127.f, fmaxf(-128.f, v));
    qr[k] = (signed char)(int)v;
  }
}

__global__ __launch_bounds__(256) void tmva_gemm(
    const signed char* __restrict__ qa, const float* __restrict__ ascale,
    const signed char* __restrict__ qw, const float* __restrict__ wslot,
    const float* __restrict__ bias, float* __restrict__ out,
    int N, int K, int ldo) {
  __shared__ signed char lsA[2][16 * 128];
  int tid = threadIdx.x;
  int lane = tid & 31, wave = tid >> 5;
  int hf = lane >> 4, l15 = lane & 15;
  int ntiles = (N + 15) >> 4;
  int tn = blockIdx.x * 8 + wave;
  if (tn >= ntiles) tn = ntiles - 1;
  long rowA = (long)blockIdx.y * 16;
  int col = tn * 16 + l15;
  int ar = tid >> 4, ac = (tid & 15) << 3;
  const signed char* gA = qa + (rowA + ar) * (long)K + ac;
  signed char* l0 = &lsA[0][ar * 128 + ac];
  signed char* l1 = &lsA[1][ar * 128 + ac];
  const signed char* wp = qw + (long)col * K + 16 * hf;
  int rb = l15 * 128 + 16 * hf;
  v8i acc = {0, 0, 0, 0, 0, 0, 0, 0};
  int iters = K >> 7;
  tmva_cp8(gA, l0);
  tmva_cpwait();
  __syncthreads();
  for (int it = 0; it < iters; ++it) {
    int p = it & 1;
    if (it + 1 < iters) tmva_cp8(gA + ((long)(it + 1) << 7), p ? l0 : l1);
    const signed char* base = lsA[p];
    int k0 = it << 7;
#pragma unroll
    for (int s = 0; s < 2; ++s) {
      int sb = rb + s * 64;
      int2 p0 = *(const int2*)&base[sb];
      int2 p1 = *(const int2*)&base[sb + 8];
      int2 p2 = *(const int2*)&base[sb + 32];
      int2 p3 = *(const int2*)&base[sb + 40];
      v8i a; a[0]=p0.x; a[1]=p0.y; a[2]=p1.x; a[3]=p1.y; a[4]=p2.x; a[5]=p2.y; a[6]=p3.x; a[7]=p3.y;
      int4 b0 = *(const int4*)(wp + k0 + s * 64);
      int4 b1 = *(const int4*)(wp + k0 + s * 64 + 32);
      v8i b; b[0]=b0.x; b[1]=b0.y; b[2]=b0.z; b[3]=b0.w; b[4]=b1.x; b[5]=b1.y; b[6]=b1.z; b[7]=b1.w;
      acc = __builtin_amdgcn_wmma_i32_16x16x64_iu8(true, a, true, b, acc, false, false);
    }
    tmva_cpwait();
    __syncthreads();
  }
  float wd = wslot[0];
  float bcol = bias ? bias[col] : 0.f;
#pragma unroll
  for (int r = 0; r < 8; ++r) {
    long row = rowA + r + 8 * hf;
    out[row * (long)ldo + col] = (float)acc[r] * ascale[row] * wd + bcol;
  }
}

__global__ __launch_bounds__(256) void tmva_gemm_glu(
    const signed char* __restrict__ qa, const float* __restrict__ ascale,
    const signed char* __restrict__ qw, const float* __restrict__ wslot,
    float* __restrict__ out, int N, int K) {
  __shared__ signed char lsA[2][16 * 128];
  int tid = threadIdx.x;
  int lane = tid & 31, wave = tid >> 5;
  int hf = lane >> 4, l15 = lane & 15;
  int ntiles = N >> 4;
  int tn = blockIdx.x * 8 + wave;
  if (tn >= ntiles) tn = ntiles - 1;
  long rowA = (long)blockIdx.y * 16;
  int col = tn * 16 + l15;
  int ar = tid >> 4, ac = (tid & 15) << 3;
  const signed char* gA = qa + (rowA + ar) * (long)K + ac;
  signed char* l0 = &lsA[0][ar * 128 + ac];
  signed char* l1 = &lsA[1][ar * 128 + ac];
  const signed char* wp  = qw + (long)col * K + 16 * hf;
  const signed char* wp2 = qw + (long)(col + N) * K + 16 * hf;
  int rb = l15 * 128 + 16 * hf;
  v8i acc  = {0, 0, 0, 0, 0, 0, 0, 0};
  v8i acc2 = {0, 0, 0, 0, 0, 0, 0, 0};
  int iters = K >> 7;
  tmva_cp8(gA, l0);
  tmva_cpwait();
  __syncthreads();
  for (int it = 0; it < iters; ++it) {
    int p = it & 1;
    if (it + 1 < iters) tmva_cp8(gA + ((long)(it + 1) << 7), p ? l0 : l1);
    const signed char* base = lsA[p];
    int k0 = it << 7;
#pragma unroll
    for (int s = 0; s < 2; ++s) {
      int sb = rb + s * 64;
      int2 p0 = *(const int2*)&base[sb];
      int2 p1 = *(const int2*)&base[sb + 8];
      int2 p2 = *(const int2*)&base[sb + 32];
      int2 p3 = *(const int2*)&base[sb + 40];
      v8i a; a[0]=p0.x; a[1]=p0.y; a[2]=p1.x; a[3]=p1.y; a[4]=p2.x; a[5]=p2.y; a[6]=p3.x; a[7]=p3.y;
      int4 b0 = *(const int4*)(wp + k0 + s * 64);
      int4 b1 = *(const int4*)(wp + k0 + s * 64 + 32);
      v8i b; b[0]=b0.x; b[1]=b0.y; b[2]=b0.z; b[3]=b0.w; b[4]=b1.x; b[5]=b1.y; b[6]=b1.z; b[7]=b1.w;
      acc = __builtin_amdgcn_wmma_i32_16x16x64_iu8(true, a, true, b, acc, false, false);
      int4 c0 = *(const int4*)(wp2 + k0 + s * 64);
      int4 c1 = *(const int4*)(wp2 + k0 + s * 64 + 32);
      v8i b2; b2[0]=c0.x; b2[1]=c0.y; b2[2]=c0.z; b2[3]=c0.w; b2[4]=c1.x; b2[5]=c1.y; b2[6]=c1.z; b2[7]=c1.w;
      acc2 = __builtin_amdgcn_wmma_i32_16x16x64_iu8(true, a, true, b2, acc2, false, false);
    }
    tmva_cpwait();
    __syncthreads();
  }
  float wd = wslot[0];
#pragma unroll
  for (int r = 0; r < 8; ++r) {
    long row = rowA + r + 8 * hf;
    float as = ascale[row] * wd;
    float g = (float)acc[r] * as;
    float y2 = (float)acc2[r] * as;
    out[row * (long)N + col] = (g / (1.f + expf(-g))) * y2;
  }
}

__global__ void tmva_patch(const float* __restrict__ x, const float* __restrict__ pw,
                           const float* __restrict__ pb, const float* __restrict__ pos,
                           float* __restrict__ xe) {
  long idx = (long)blockIdx.x * 256 + threadIdx.x;
  if (idx >= (long)NTOK * HDIM) return;
  int h = (int)(idx % HDIM);
  long tok = idx / HDIM;
  int bn = (int)(tok >> 8), l = (int)(tok & 255);
  int py = l >> 4, px = l & 15;
  float s = pb[h];
  const float* wrow = pw + (long)h * 16;
#pragma unroll
  for (int k = 0; k < 16; ++k) {
    int cin = k >> 2, dy = (k >> 1) & 1, dx = k & 1;
    s += x[(((long)bn * 4 + cin) * 32 + py * 2 + dy) * 32 + px * 2 + dx] * wrow[k];
  }
  xe[idx] = s + pos[(long)l * HDIM + h];
}

__global__ void tmva_tstep(const float* __restrict__ t, float* __restrict__ o) {
  int idx = blockIdx.x * 256 + threadIdx.x;
  if (idx >= BNIMG * 256) return;
  int n = idx >> 8, j = idx & 255;
  int hx = j & 127;
  float freq = expf(-9.210340371976184f * (float)hx / 128.f);
  float a = t[n] * freq;
  o[idx] = (j < 128) ? cosf(a) : sinf(a);
}

__global__ void tmva_pool(const float* __restrict__ ehs, float* __restrict__ pm) {
  int idx = blockIdx.x * 256 + threadIdx.x;
  if (idx >= BNIMG * 768) return;
  int n = idx / 768, d = idx % 768;
  int b = n >> 2;
  float s = 0.f;
  for (int i = 0; i < 77; ++i) s += ehs[((long)b * 77 + i) * 768 + d];
  pm[idx] = s / 77.f;
}

__global__ void tmva_silu(const float* __restrict__ x, float* __restrict__ y, long n) {
  long i = (long)blockIdx.x * 256 + threadIdx.x;
  if (i >= n) return;
  float v = x[i];
  y[i] = v / (1.f + expf(-v));
}

__global__ void tmva_add(float* __restrict__ a, const float* __restrict__ b, long n) {
  long i = (long)blockIdx.x * 256 + threadIdx.x;
  if (i >= n) return;
  a[i] += b[i];
}

__global__ void tmva_lnmod(const float* __restrict__ x, float* __restrict__ o,
                           const float* __restrict__ ada, int stride, int scOff, int shOff) {
  int r = blockIdx.x;
  const float* xr = x + (long)r * HDIM;
  __shared__ float s1[256];
  __shared__ float s2[256];
  float a = 0.f, b = 0.f;
  for (int k = threadIdx.x; k < HDIM; k += 256) { float v = xr[k]; a += v; b += v * v; }
  s1[threadIdx.x] = a; s2[threadIdx.x] = b; __syncthreads();
  for (int t = 128; t > 0; t >>= 1) {
    if ((int)threadIdx.x < t) { s1[threadIdx.x] += s1[threadIdx.x + t]; s2[threadIdx.x] += s2[threadIdx.x + t]; }
    __syncthreads();
  }
  float mean = s1[0] / (float)HDIM;
  float var = s2[0] / (float)HDIM - mean * mean;
  float rs = rsqrtf(var + 1e-6f);
  int img = r >> 8;
  const float* sc = ada + (long)img * stride + scOff;
  const float* sh = ada + (long)img * stride + shOff;
  float* orow = o + (long)r * HDIM;
  for (int k = threadIdx.x; k < HDIM; k += 256)
    orow[k] = (xr[k] - mean) * rs * (1.f + sc[k]) + sh[k];
}

__global__ void tmva_gates(float* __restrict__ bi, float* __restrict__ bf, long n) {
  long i = (long)blockIdx.x * 256 + threadIdx.x;
  if (i >= n) return;
  float f = 1.f / (1.f + expf(-bf[i]));
  float v = bi[i];
  v = v / (1.f + expf(-v));
  bf[i] = f;
  bi[i] = v * (1.f - f);
}

__global__ void tmva_scan_sa(float* __restrict__ bi, const float* __restrict__ bf) {
  int idx = blockIdx.x * 256 + threadIdx.x;
  if (idx >= BNIMG * HDIM) return;
  int b = idx / HDIM, d = idx % HDIM;
  float h = 0.f;
  for (int l = 0; l < LTOK; ++l) {
    long o = ((long)(b * LTOK + l)) * HDIM + d;
    h = bf[o] * h + bi[o];
    bi[o] = h;
  }
}

__global__ void tmva_scan_mv(float* __restrict__ bi, const float* __restrict__ bf) {
  long idx = (long)blockIdx.x * 256 + threadIdx.x;
  if (idx >= (long)(BNIMG / NVIEW) * LTOK * HDIM) return;
  long bl = idx / HDIM;
  int d = (int)(idx % HDIM);
  float h = 0.f;
  for (int v = 0; v < NVIEW; ++v) {
    long o = (bl * NVIEW + v) * HDIM + d;
    h = bf[o] * h + bi[o];
    bi[o] = h;
  }
}

__global__ void tmva_hgate(float* __restrict__ o, const float* __restrict__ g,
                           const float* __restrict__ gn) {
  int idx = blockIdx.x * 256 + threadIdx.x;
  if (idx >= NTOK * NHEAD) return;
  long base = (long)(idx / NHEAD) * HDIM + (idx % NHEAD) * HEADD;
  float s = 0.f;
  for (int j = 0; j < HEADD; ++j) { float v = o[base + j]; s += v * v; }
  float rs = rsqrtf(s / (float)HEADD + 1e-5f);
  for (int j = 0; j < HEADD; ++j) {
    float gv = g[base + j];
    o[base + j] = o[base + j] * rs * gn[j] * (gv / (1.f + expf(-gv)));
  }
}

__global__ void tmva_resid(float* __restrict__ xe, const float* __restrict__ s1,
                           const float* __restrict__ s2, const float* __restrict__ ada,
                           int stride, int gOff) {
  long idx = (long)blockIdx.x * 256 + threadIdx.x;
  if (idx >= (long)NTOK * HDIM) return;
  long r = idx / HDIM;
  int d = (int)(idx % HDIM);
  int img = (int)(r >> 8);
  float g = ada[(long)img * stride + gOff + d];
  float v = s1[idx];
  if (s2) {
    int b = (int)(r >> 10);
    int vv = (int)((r >> 8) & 3);
    int l = (int)(r & 255);
    long pr = ((long)(b * LTOK + l)) * NVIEW + vv;
    v += s2[pr * HDIM + d];
  }
  xe[idx] += g * v;
}

__global__ void tmva_unpatch(const float* __restrict__ tok, float* __restrict__ out) {
  int idx = blockIdx.x * 256 + threadIdx.x;
  if (idx >= BNIMG * 8 * 32 * 32) return;
  int n = idx >> 13;
  int rem = idx & 8191;
  int c = rem >> 10;
  int hw = rem & 1023;
  int hh = hw >> 5, ww = hw & 31;
  int py = hh >> 1, dy = hh & 1, px = ww >> 1, dx = ww & 1;
  long row = (long)n * LTOK + py * 16 + px;
  out[idx] = tok[row * 32 + (dy * 2 + dx) * 8 + c];
}

extern "C" void kernel_launch(void* const* d_in, const int* in_sizes, int n_in,
                              void* d_out, int out_size, void* d_ws, size_t ws_size,
                              hipStream_t stream) {
  (void)in_sizes; (void)n_in; (void)out_size;
  const float* x        = (const float*)d_in[0];
  const float* t        = (const float*)d_in[1];
  const float* ehs      = (const float*)d_in[2];
  const float* patch_w  = (const float*)d_in[4];
  const float* patch_b  = (const float*)d_in[5];
  const float* pos      = (const float*)d_in[6];
  const float* t_w1     = (const float*)d_in[7];
  const float* t_b1     = (const float*)d_in[8];
  const float* t_w2     = (const float*)d_in[9];
  const float* t_b2     = (const float*)d_in[10];
  const float* prompt_w = (const float*)d_in[11];
  const float* prompt_b = (const float*)d_in[12];
  const float* adaln_w  = (const float*)d_in[13];
  const float* adaln_b  = (const float*)d_in[14];
  const float* sa_iw    = (const float*)d_in[15];
  const float* sa_fw    = (const float*)d_in[16];
  const float* sa_gw    = (const float*)d_in[17];
  const float* sa_ow    = (const float*)d_in[18];
  const float* sa_gn    = (const float*)d_in[19];
  const float* mv_iw    = (const float*)d_in[20];
  const float* mv_fw    = (const float*)d_in[21];
  const float* mv_gw    = (const float*)d_in[22];
  const float* mv_ow    = (const float*)d_in[23];
  const float* mv_gn    = (const float*)d_in[24];
  const float* mlp_gw   = (const float*)d_in[25];
  const float* mlp_dw   = (const float*)d_in[26];
  const float* fin_w    = (const float*)d_in[27];
  const float* fin_b    = (const float*)d_in[28];
  const float* fin_aw   = (const float*)d_in[29];
  const float* fin_ab   = (const float*)d_in[30];

  char* W = (char*)d_ws;
  size_t off = 0;
  auto alloc = [&](size_t sz) { size_t r = off; off += (sz + 255) & ~(size_t)255; return r; };
  float*       XE     = (float*)(W + alloc((size_t)NTOK * HDIM * 4));
  float*       XM     = (float*)(W + alloc((size_t)NTOK * HDIM * 4));
  signed char* QA     = (signed char*)(W + alloc((size_t)NTOK * INTERD));
  signed char* QW     = (signed char*)(W + alloc((size_t)(2 * INTERD) * HDIM));
  float*       ASC    = (float*)(W + alloc((size_t)NTOK * 4));
  float*       WSLOT  = (float*)(W + alloc(256));
  float*       PART   = (float*)(W + alloc(4096));
  float*       TFREQ  = (float*)(W + alloc((size_t)BNIMG * 256 * 4));
  float*       TE     = (float*)(W + alloc((size_t)BNIMG * HDIM * 4));
  float*       CBUF   = (float*)(W + alloc((size_t)BNIMG * HDIM * 4));
  float*       SCOND  = (float*)(W + alloc((size_t)BNIMG * HDIM * 4));
  float*       PM     = (float*)(W + alloc((size_t)BNIMG * 768 * 4));
  float*       ADA    = (float*)(W + alloc((size_t)BNIMG * 6 * HDIM * 4));
  float*       FINADA = (float*)(W + alloc((size_t)BNIMG * 2 * HDIM * 4));
  float*       BI     = (float*)(W + alloc((size_t)NTOK * HDIM * 4));
  float*       BF     = (float*)(W + alloc((size_t)NTOK * HDIM * 4));
  float*       BG     = (float*)(W + alloc((size_t)NTOK * HDIM * 4));
  float*       BM     = (float*)(W + alloc((size_t)NTOK * HDIM * 4));
  float*       BS     = (float*)(W + alloc((size_t)NTOK * HDIM * 4));
  float*       HB     = BI;
  float*       TOK    = BI;
  if (ws_size < off) return;

  auto cdiv = [](long a, long b) { return (a + b - 1) / b; };
  auto wq = [&](const float* w, long n) {
    tmva_wred1<<<512, 256, 0, stream>>>(w, n, PART);
    tmva_wred2<<<1, 256, 0, stream>>>(PART, 512, n, WSLOT);
    tmva_wquant<<<(unsigned)cdiv(n, 2048), 256, 0, stream>>>(w, QW, n, WSLOT);
  };
  auto aq = [&](const float* xx, int M, int K, int perm) {
    tmva_aquant<<<(unsigned)M, 256, 0, stream>>>(xx, QA, ASC, K, perm);
  };
  auto gemm = [&](int M, int N, int K, const float* bias, float* out) {
    dim3 g((unsigned)cdiv((N + 15) / 16, 8), (unsigned)(M / 16));
    tmva_gemm<<<g, 256, 0, stream>>>(QA, ASC, QW, WSLOT, bias, out, N, K, N);
  };
  auto bl = [&](const float* xx, int M, int K, const float* w, int N,
                const float* bias, float* out, int perm) {
    wq(w, (long)N * K);
    aq(xx, M, K, perm);
    gemm(M, N, K, bias, out);
  };
  auto ew = [&](long n) { return (unsigned)cdiv(n, 256); };

  const long NH_ = (long)NTOK * HDIM;

  tmva_patch<<<ew(NH_), 256, 0, stream>>>(x, patch_w, patch_b, pos, XE);
  tmva_tstep<<<ew(BNIMG * 256), 256, 0, stream>>>(t, TFREQ);
  bl(TFREQ, BNIMG, 256, t_w1, HDIM, t_b1, TE, 0);
  tmva_silu<<<ew((long)BNIMG * HDIM), 256, 0, stream>>>(TE, TE, (long)BNIMG * HDIM);
  bl(TE, BNIMG, HDIM, t_w2, HDIM, t_b2, CBUF, 0);
  tmva_pool<<<ew(BNIMG * 768), 256, 0, stream>>>(ehs, PM);
  bl(PM, BNIMG, 768, prompt_w, HDIM, prompt_b, TE, 0);
  tmva_add<<<ew((long)BNIMG * HDIM), 256, 0, stream>>>(CBUF, TE, (long)BNIMG * HDIM);

  for (int i = 0; i < 4; ++i) {
    const long wij = (long)i * HDIM * HDIM;
    tmva_silu<<<ew((long)BNIMG * HDIM), 256, 0, stream>>>(CBUF, SCOND, (long)BNIMG * HDIM);
    bl(SCOND, BNIMG, HDIM, adaln_w + (long)i * 6 * HDIM * HDIM, 6 * HDIM,
       adaln_b + (long)i * 6 * HDIM, ADA, 0);

    tmva_lnmod<<<NTOK, 256, 0, stream>>>(XE, XM, ADA, 6 * HDIM, HDIM, 0);

    aq(XM, NTOK, HDIM, 0);
    wq(sa_iw + wij, (long)HDIM * HDIM); gemm(NTOK, HDIM, HDIM, nullptr, BI);
    wq(sa_fw + wij, (long)HDIM * HDIM); gemm(NTOK, HDIM, HDIM, nullptr, BF);
    wq(sa_gw + wij, (long)HDIM * HDIM); gemm(NTOK, HDIM, HDIM, nullptr, BG);
    tmva_gates<<<ew(NH_), 256, 0, stream>>>(BI, BF, NH_);
    tmva_scan_sa<<<ew(BNIMG * HDIM), 256, 0, stream>>>(BI, BF);
    tmva_hgate<<<ew(NTOK * NHEAD), 256, 0, stream>>>(BI, BG, sa_gn + i * HEADD);
    bl(BI, NTOK, HDIM, sa_ow + wij, HDIM, nullptr, BS, 0);

    aq(XM, NTOK, HDIM, 1);
    wq(mv_iw + wij, (long)HDIM * HDIM); gemm(NTOK, HDIM, HDIM, nullptr, BI);
    wq(mv_fw + wij, (long)HDIM * HDIM); gemm(NTOK, HDIM, HDIM, nullptr, BF);
    wq(mv_gw + wij, (long)HDIM * HDIM); gemm(NTOK, HDIM, HDIM, nullptr, BG);
    tmva_gates<<<ew(NH_), 256, 0, stream>>>(BI, BF, NH_);
    tmva_scan_mv<<<ew((long)1024 * HDIM), 256, 0, stream>>>(BI, BF);
    tmva_hgate<<<ew(NTOK * NHEAD), 256, 0, stream>>>(BI, BG, mv_gn + i * HEADD);
    bl(BI, NTOK, HDIM, mv_ow + wij, HDIM, nullptr, BM, 0);

    tmva_resid<<<ew(NH_), 256, 0, stream>>>(XE, BS, BM, ADA, 6 * HDIM, 2 * HDIM);

    tmva_lnmod<<<NTOK, 256, 0, stream>>>(XE, XM, ADA, 6 * HDIM, 4 * HDIM, 3 * HDIM);
    wq(mlp_gw + (long)i * 2 * INTERD * HDIM, (long)2 * INTERD * HDIM);
    aq(XM, NTOK, HDIM, 0);
    {
      dim3 g((unsigned)((INTERD / 16) / 8), (unsigned)(NTOK / 16));
      tmva_gemm_glu<<<g, 256, 0, stream>>>(QA, ASC, QW, WSLOT, HB, INTERD, HDIM);
    }
    bl(HB, NTOK, INTERD, mlp_dw + (long)i * HDIM * INTERD, HDIM, nullptr, BS, 0);
    tmva_resid<<<ew(NH_), 256, 0, stream>>>(XE, BS, nullptr, ADA, 6 * HDIM, 5 * HDIM);
  }

  tmva_silu<<<ew((long)BNIMG * HDIM), 256, 0, stream>>>(CBUF, SCOND, (long)BNIMG * HDIM);
  bl(SCOND, BNIMG, HDIM, fin_aw, 2 * HDIM, fin_ab, FINADA, 0);
  tmva_lnmod<<<NTOK, 256, 0, stream>>>(XE, XM, FINADA, 2 * HDIM, HDIM, 0);
  bl(XM, NTOK, HDIM, fin_w, 32, fin_b, TOK, 0);
  tmva_unpatch<<<ew(BNIMG * 8 * 32 * 32), 256, 0, stream>>>(TOK, (float*)d_out);
}
